// CausalSelfAttention_84241488544381
// MI455X (gfx1250) — matrix operand order repaired and checked by reading, not compiled, not run on hardware
//
#include <hip/hip_runtime.h>
#include <hip/hip_bf16.h>

typedef _Float16 h16;
typedef __attribute__((ext_vector_type(16))) _Float16 v16h;
typedef __attribute__((ext_vector_type(8)))  _Float16 v8h;
typedef __attribute__((ext_vector_type(8)))  float    v8f;

#define TOK   2048
#define DIM   1024
#define NHEAD 16
#define HDIM  64
#define E3    3072
#define EPSV  1e-4f
#define RMSEPS 1.1920929e-07f
#define ASCALE 0.12f
#define NEGINF (-1e30f)

// ---------------------------------------------------------------------------
// WMMA operand loaders — k-contiguous layouts (this code's choice) for V_WMMA_F32_16X16X32_F16 (wave32)
// A (16x32 f16): lane L row M=L&15; halves 0..15 -> K = 16*(L>=16)+0..15
//                (contiguous, matches B)                 => two 16B loads
// B (32x16 f16): lane L col N=L&15; K chunk = 16*(L>=16)+0..15 => one 32B load
// C/D (16x16 f32): elem r of lane L is (M = r + 8*(L>=16), N = L&15)
// ---------------------------------------------------------------------------
__device__ __forceinline__ v16h load_A16x32(const h16* base, int ld, int lane) {
  int m  = lane & 15;
  int kb = (lane >> 4) << 4;
  const h16* p = base + m * ld + kb;
  union { v16h v; v8h h[2]; } u;
  u.h[0] = *(const v8h*)(p);
  u.h[1] = *(const v8h*)(p + 8);
  return u.v;
}

__device__ __forceinline__ v16h load_B32x16(const h16* base, int ld, int lane) {
  int n  = lane & 15;
  int kc = (lane >> 4) << 4;
  return *(const v16h*)(base + n * ld + kc);
}

// ---------------------------------------------------------------------------
// K1: f32 -> f16 convert (weights)
// ---------------------------------------------------------------------------
__global__ void __launch_bounds__(256) cvt_f32_f16(const float* __restrict__ s,
                                                   h16* __restrict__ d, int n) {
  int i = blockIdx.x * blockDim.x + threadIdx.x;
  int st = gridDim.x * blockDim.x;
  for (; i < n; i += st) d[i] = (h16)s[i];
}

// ---------------------------------------------------------------------------
// block reduction helper (256 threads = 8 waves)
// ---------------------------------------------------------------------------
__device__ __forceinline__ float block_sum(float v, float* sbuf) {
  #pragma unroll
  for (int off = 1; off < 32; off <<= 1) v += __shfl_xor(v, off, 32);
  int lane = threadIdx.x & 31, wave = threadIdx.x >> 5;
  if (lane == 0) sbuf[wave] = v;
  __syncthreads();
  float tot = 0.f;
  #pragma unroll
  for (int i = 0; i < 8; ++i) tot += sbuf[i];
  __syncthreads();
  return tot;
}

// ---------------------------------------------------------------------------
// K2: ref = shift(x); x_hyp = logmap(ref, x, c=1); write ref (fp32 output #2)
// ---------------------------------------------------------------------------
__global__ void __launch_bounds__(256)
logmap_kernel(const float* __restrict__ x, h16* __restrict__ xhyp,
              float* __restrict__ ref_out) {
  __shared__ float sb[8];
  const float c = 1.0f;
  int t  = blockIdx.x;
  int e0 = threadIdx.x * 4;
  float4 uv = *(const float4*)(x + t * DIM + e0);
  float4 rv = make_float4(0.f, 0.f, 0.f, 0.f);
  if (t > 0) rv = *(const float4*)(x + (t - 1) * DIM + e0);
  *(float4*)(ref_out + t * DIM + e0) = rv;

  float uu[4] = {uv.x, uv.y, uv.z, uv.w};
  float rr[4] = {rv.x, rv.y, rv.z, rv.w};
  float pxn = 0.f, pun = 0.f, pip = 0.f;
  #pragma unroll
  for (int i = 0; i < 4; ++i) {
    pxn += rr[i] * rr[i];
    pun += uu[i] * uu[i];
    pip += rr[i] * uu[i];
  }
  float xn2 = block_sum(pxn, sb);
  float un2 = block_sum(pun, sb);
  float mip = -block_sum(pip, sb);

  float coefA = 1.f + 2.f * c * mip + c * un2;
  float coefB = 1.f - c * xn2;
  float invd  = 1.f / fmaxf(1.f + 2.f * c * mip + c * c * xn2 * un2, EPSV);
  float mob[4]; float pmn = 0.f;
  #pragma unroll
  for (int i = 0; i < 4; ++i) {
    mob[i] = (coefA * (-rr[i]) + coefB * uu[i]) * invd;
    pmn   += mob[i] * mob[i];
  }
  float mn = sqrtf(block_sum(pmn, sb));
  float sf = 2.f / (1.f + c * xn2);
  float cf = 2.f / (sf * sqrtf(c) + EPSV);
  float arg = fminf(fmaxf(sqrtf(c * mn), -0.99f), 0.99f);
  float ath = 0.5f * __logf((1.f + arg) / (1.f - arg));
  float scl = cf * ath / (mn + EPSV);
  h16* o = xhyp + t * DIM + e0;
  #pragma unroll
  for (int i = 0; i < 4; ++i) o[i] = (h16)(scl * mob[i]);
}

// ---------------------------------------------------------------------------
// K3: C[M,N] f32 = A[M,K] f16 x B[N,K]ᵀ f16.
// Block = 16 rows x 512 cols (8 waves x 4 n-tiles). The block-shared 16x32 A
// tile is staged to LDS with global_load_async_to_lds_b128 (ASYNCcnt),
// double-buffered so the async DMA for step s+1 overlaps B-loads + 4 WMMAs
// of step s. One barrier per step.
// ---------------------------------------------------------------------------
__global__ void __launch_bounds__(256)
gemm_wmma_f16f32(const h16* __restrict__ A, const h16* __restrict__ Bw,
                 float* __restrict__ C, int M, int N, int K) {
  __shared__ h16 aBuf[2][16 * 32];
  int tid  = threadIdx.x;
  int lane = tid & 31, wave = tid >> 5;
  int ngrp = N >> 9;                       // 512 cols per block
  int mt   = blockIdx.x / ngrp;
  int grp  = blockIdx.x % ngrp;
  int row0 = mt << 4;
  int col0 = (grp * 8 + wave) << 6;        // 64 cols per wave
  int m    = lane & 15;
  int hi   = lane >> 4;

  // loader threads (waves 0-1): one 16B async chunk each per step
  int arow = tid >> 2, aq = tid & 3;       // row 0..15, 16B chunk 0..3
  const h16* aSrc = A + (row0 + arow) * K + aq * 8;
  unsigned aDst0 = (unsigned)(size_t)&aBuf[0][arow * 32 + aq * 8];
  unsigned aDst1 = (unsigned)(size_t)&aBuf[1][arow * 32 + aq * 8];

  const h16* Bb = Bw + col0 * K;
  v8f acc0 = {}, acc1 = {}, acc2 = {}, acc3 = {};

  if (tid < 64) {                          // preload buffer 0
    unsigned long long ga = (unsigned long long)(size_t)aSrc;
    asm volatile("global_load_async_to_lds_b128 %0, %1, off"
                 :: "v"(aDst0), "v"(ga) : "memory");
  }
  asm volatile("s_wait_asynccnt 0x0" ::: "memory");
  __syncthreads();

  int nstep = K >> 5;
  for (int s = 0; s < nstep; ++s) {
    int k0  = s << 5;
    int buf = s & 1;
    // kick off async copy of the NEXT A tile (overlaps compute below)
    if (s + 1 < nstep && tid < 64) {
      unsigned long long ga = (unsigned long long)(size_t)(aSrc + k0 + 32);
      unsigned dst = buf ? aDst0 : aDst1;
      asm volatile("global_load_async_to_lds_b128 %0, %1, off"
                   :: "v"(dst), "v"(ga) : "memory");
    }
    // A operand from LDS (two ds_load_b128 in A layout)
    union { v16h v; v8h h[2]; } ua;
    const h16* ap = &aBuf[buf][m * 32 + (hi << 4)];
    ua.h[0] = *(const v8h*)(ap);
    ua.h[1] = *(const v8h*)(ap + 8);
    // B operands (4 n-tiles, each one 32B global load per lane)
    v16h b0 = load_B32x16(Bb + 0 * 16 * K + k0, K, lane);
    v16h b1 = load_B32x16(Bb + 1 * 16 * K + k0, K, lane);
    v16h b2 = load_B32x16(Bb + 2 * 16 * K + k0, K, lane);
    v16h b3 = load_B32x16(Bb + 3 * 16 * K + k0, K, lane);
    acc0 = __builtin_amdgcn_wmma_f32_16x16x32_f16(false, ua.v, false, b0,
                                                  (short)0, acc0, false, false);
    acc1 = __builtin_amdgcn_wmma_f32_16x16x32_f16(false, ua.v, false, b1,
                                                  (short)0, acc1, false, false);
    acc2 = __builtin_amdgcn_wmma_f32_16x16x32_f16(false, ua.v, false, b2,
                                                  (short)0, acc2, false, false);
    acc3 = __builtin_amdgcn_wmma_f32_16x16x32_f16(false, ua.v, false, b3,
                                                  (short)0, acc3, false, false);
    // next tile landed + this tile's reads done -> safe to flip buffers
    asm volatile("s_wait_asynccnt 0x0" ::: "memory");
    __syncthreads();
  }

  int mb = hi << 3;
  float* Cr = C + (row0 + mb) * N + col0 + m;
  #pragma unroll
  for (int r = 0; r < 8; ++r) {
    Cr[r * N + 0]  = acc0[r];
    Cr[r * N + 16] = acc1[r];
    Cr[r * N + 32] = acc2[r];
    Cr[r * N + 48] = acc3[r];
  }
}

// ---------------------------------------------------------------------------
// K4: per (token, head): rms_norm + rotary for q,k -> f16; v -> transposed f16
// ---------------------------------------------------------------------------
__global__ void __launch_bounds__(256)
qkv_post_kernel(const float* __restrict__ qkv, h16* __restrict__ q16,
                h16* __restrict__ k16, h16* __restrict__ vT16) {
  int t    = blockIdx.x;
  int head = threadIdx.x >> 4;
  int j    = threadIdx.x & 15;
  int p0   = 2 * j;
  float fr[2], cs[2], sn[2];
  #pragma unroll
  for (int q = 0; q < 2; ++q) {
    int p = p0 + q;
    fr[q] = (p < 16) ? __powf(1.0f / 1024.0f, (float)p * (1.0f / 15.0f)) : 0.0f;
    float th = (float)t * fr[q];
    __sincosf(th, &sn[q], &cs[q]);
  }
  #pragma unroll
  for (int e = 0; e < 2; ++e) {
    const float* src = qkv + t * E3 + e * DIM + head * HDIM;
    float x1[2], x2[2], ss = 0.f;
    #pragma unroll
    for (int q = 0; q < 2; ++q) {
      x1[q] = src[p0 + q];
      x2[q] = src[p0 + q + 32];
      ss += x1[q] * x1[q] + x2[q] * x2[q];
    }
    ss += __shfl_xor(ss, 1, 32); ss += __shfl_xor(ss, 2, 32);
    ss += __shfl_xor(ss, 4, 32); ss += __shfl_xor(ss, 8, 32);
    float rn = rsqrtf(ss * (1.0f / HDIM) + RMSEPS);
    h16* dst = (e == 0 ? q16 : k16) + t * DIM + head * HDIM;
    #pragma unroll
    for (int q = 0; q < 2; ++q) {
      float a = x1[q] * rn, b = x2[q] * rn;
      dst[p0 + q]      = (h16)( a * cs[q] + b * sn[q]);
      dst[p0 + q + 32] = (h16)(-a * sn[q] + b * cs[q]);
    }
  }
  const float* vs = qkv + t * E3 + 2 * DIM + head * HDIM;
  #pragma unroll
  for (int q = 0; q < 2; ++q) {
    vT16[(head * HDIM + p0 + q) * TOK + t]      = (h16)vs[p0 + q];
    vT16[(head * HDIM + p0 + q + 32) * TOK + t] = (h16)vs[p0 + q + 32];
  }
}

// ---------------------------------------------------------------------------
// K5: flash attention. 1 wave per (head, 16-row q-tile); key blocks of 32.
// ---------------------------------------------------------------------------
__global__ void __launch_bounds__(256)
flash_attn_kernel(const h16* __restrict__ Q, const h16* __restrict__ Km,
                  const h16* __restrict__ Vt, h16* __restrict__ Yo) {
  __shared__ h16 plds[8][16 * 32];
  int lane = threadIdx.x & 31;
  int wave = threadIdx.x >> 5;
  int gw   = blockIdx.x * 8 + wave;
  int head = gw >> 7;
  int qt   = gw & 127;
  int m    = lane & 15;
  int hi   = lane >> 4;
  h16* pt  = plds[wave];

  const h16* Qb = Q + (qt * 16) * DIM + head * HDIM;
  v16h qa0 = load_A16x32(Qb,      DIM, lane);
  v16h qa1 = load_A16x32(Qb + 32, DIM, lane);

  float m8[8], l8[8];
  v8f o0 = {}, o1 = {}, o2 = {}, o3 = {};
  #pragma unroll
  for (int r = 0; r < 8; ++r) { m8[r] = NEGINF; l8[r] = 0.f; }

  int nKB = (qt >> 1) + 1;
  for (int kb = 0; kb < nKB; ++kb) {
    v8f s[2];
    #pragma unroll
    for (int nh = 0; nh < 2; ++nh) {
      int kk = kb * 32 + nh * 16 + m;
      const h16* Kb = Km + kk * DIM + head * HDIM + hi * 16;
      v16h b0 = *(const v16h*)(Kb);
      v16h b1 = *(const v16h*)(Kb + 32);
      v8f sa = {};
      sa = __builtin_amdgcn_wmma_f32_16x16x32_f16(false, qa0, false, b0,
                                                  (short)0, sa, false, false);
      sa = __builtin_amdgcn_wmma_f32_16x16x32_f16(false, qa1, false, b1,
                                                  (short)0, sa, false, false);
      s[nh] = sa;
    }
    #pragma unroll
    for (int nh = 0; nh < 2; ++nh) {
      int kcol = kb * 32 + nh * 16 + m;
      #pragma unroll
      for (int r = 0; r < 8; ++r) {
        int qr = qt * 16 + r + hi * 8;
        float v = s[nh][r] * ASCALE;
        s[nh][r] = (kcol <= qr) ? v : NEGINF;
      }
    }
    #pragma unroll
    for (int r = 0; r < 8; ++r) {
      float rm = fmaxf(s[0][r], s[1][r]);
      rm = fmaxf(rm, __shfl_xor(rm, 1, 32));
      rm = fmaxf(rm, __shfl_xor(rm, 2, 32));
      rm = fmaxf(rm, __shfl_xor(rm, 4, 32));
      rm = fmaxf(rm, __shfl_xor(rm, 8, 32));
      float nm = fmaxf(m8[r], rm);
      float al = __expf(m8[r] - nm);
      float p0v = __expf(s[0][r] - nm);
      float p1v = __expf(s[1][r] - nm);
      float rs = p0v + p1v;
      rs += __shfl_xor(rs, 1, 32); rs += __shfl_xor(rs, 2, 32);
      rs += __shfl_xor(rs, 4, 32); rs += __shfl_xor(rs, 8, 32);
      l8[r] = l8[r] * al + rs;
      m8[r] = nm;
      o0[r] *= al; o1[r] *= al; o2[r] *= al; o3[r] *= al;
      pt[(r + hi * 8) * 32 + m]      = (h16)p0v;
      pt[(r + hi * 8) * 32 + 16 + m] = (h16)p1v;
    }
    asm volatile("s_wait_dscnt 0x0" ::: "memory");
    v16h pa;
    {
      union { v16h v; v8h h[2]; } u;
      const h16* pp = pt + m * 32 + (hi << 4);
      u.h[0] = *(const v8h*)(pp);
      u.h[1] = *(const v8h*)(pp + 8);
      pa = u.v;
    }
    const h16* Vb = Vt + head * HDIM * TOK + kb * 32 + hi * 16;
    v16h vb0 = *(const v16h*)(Vb + ( 0 + m) * TOK);
    v16h vb1 = *(const v16h*)(Vb + (16 + m) * TOK);
    v16h vb2 = *(const v16h*)(Vb + (32 + m) * TOK);
    v16h vb3 = *(const v16h*)(Vb + (48 + m) * TOK);
    o0 = __builtin_amdgcn_wmma_f32_16x16x32_f16(false, pa, false, vb0,
                                                (short)0, o0, false, false);
    o1 = __builtin_amdgcn_wmma_f32_16x16x32_f16(false, pa, false, vb1,
                                                (short)0, o1, false, false);
    o2 = __builtin_amdgcn_wmma_f32_16x16x32_f16(false, pa, false, vb2,
                                                (short)0, o2, false, false);
    o3 = __builtin_amdgcn_wmma_f32_16x16x32_f16(false, pa, false, vb3,
                                                (short)0, o3, false, false);
  }
  #pragma unroll
  for (int r = 0; r < 8; ++r) {
    float inv = 1.0f / l8[r];
    int row = qt * 16 + r + hi * 8;
    h16* yp = Yo + row * DIM + head * HDIM + m;
    yp[0]  = (h16)(o0[r] * inv);
    yp[16] = (h16)(o1[r] * inv);
    yp[32] = (h16)(o2[r] * inv);
    yp[48] = (h16)(o3[r] * inv);
  }
}

// ---------------------------------------------------------------------------
// K6: out = expmap(ref, proj, c=1)
// ---------------------------------------------------------------------------
__global__ void __launch_bounds__(256)
expmap_kernel(const float* __restrict__ proj, const float* __restrict__ ref,
              float* __restrict__ out) {
  __shared__ float sb[8];
  const float c = 1.0f;
  int t  = blockIdx.x;
  int e0 = threadIdx.x * 4;
  float4 xv4 = *(const float4*)(ref  + t * DIM + e0);
  float4 vv4 = *(const float4*)(proj + t * DIM + e0);
  float xv[4] = {xv4.x, xv4.y, xv4.z, xv4.w};
  float vv[4] = {vv4.x, vv4.y, vv4.z, vv4.w};
  float pxn = 0.f, pvn = 0.f, pip = 0.f;
  #pragma unroll
  for (int i = 0; i < 4; ++i) {
    pxn += xv[i] * xv[i];
    pvn += vv[i] * vv[i];
    pip += xv[i] * vv[i];
  }
  float xn2 = block_sum(pxn, sb);
  float vn2 = block_sum(pvn, sb);
  float ipv = block_sum(pip, sb);

  float sf  = 2.f / (1.f + c * xn2);
  float vn  = sqrtf(vn2);
  float arg = sqrtf(fminf(fmaxf(c * sf * vn2 * 0.5f, 0.f), 8.f));
  float s2  = (1.f / sqrtf(c)) * tanhf(arg) / (vn + EPSV);
  float ip2 = s2 * ipv;
  float sn2 = s2 * s2 * vn2;
  float coefA = 1.f + 2.f * c * ip2 + c * sn2;
  float coefB = (1.f - c * xn2) * s2;
  float invd  = 1.f / fmaxf(1.f + 2.f * c * ip2 + c * c * xn2 * sn2, EPSV);
  float* o = out + t * DIM + e0;
  #pragma unroll
  for (int i = 0; i < 4; ++i) o[i] = (coefA * xv[i] + coefB * vv[i]) * invd;
}

// ---------------------------------------------------------------------------
// workspace layout (bytes)
// ---------------------------------------------------------------------------
static const size_t OFF_XHYP  = 0;
static const size_t OFF_QKVW  = OFF_XHYP + (size_t)TOK * DIM * 2;
static const size_t OFF_CPW   = OFF_QKVW + (size_t)E3 * DIM * 2;
static const size_t OFF_QKV   = OFF_CPW  + (size_t)DIM * DIM * 2;
static const size_t OFF_Q16   = OFF_QKV  + (size_t)TOK * E3 * 4;
static const size_t OFF_K16   = OFF_Q16  + (size_t)TOK * DIM * 2;
static const size_t OFF_VT16  = OFF_K16  + (size_t)TOK * DIM * 2;
static const size_t OFF_Y16   = OFF_VT16 + (size_t)TOK * DIM * 2;
static const size_t OFF_PROJ  = OFF_Y16  + (size_t)TOK * DIM * 2;

extern "C" void kernel_launch(void* const* d_in, const int* in_sizes, int n_in,
                              void* d_out, int out_size, void* d_ws, size_t ws_size,
                              hipStream_t stream) {
  const float* x        = (const float*)d_in[0];
  const float* qkv_w    = (const float*)d_in[1];
  const float* c_proj_w = (const float*)d_in[2];
  float* y_out   = (float*)d_out;
  float* ref_out = y_out + (size_t)TOK * DIM;

  char* ws = (char*)d_ws;
  h16*   xhyp   = (h16*)(ws + OFF_XHYP);
  h16*   qkvw16 = (h16*)(ws + OFF_QKVW);
  h16*   cpw16  = (h16*)(ws + OFF_CPW);
  float* qkvf   = (float*)(ws + OFF_QKV);
  h16*   q16    = (h16*)(ws + OFF_Q16);
  h16*   k16    = (h16*)(ws + OFF_K16);
  h16*   vT16   = (h16*)(ws + OFF_VT16);
  h16*   y16    = (h16*)(ws + OFF_Y16);
  float* projf  = (float*)(ws + OFF_PROJ);

  cvt_f32_f16<<<1024, 256, 0, stream>>>(qkv_w,    qkvw16, E3 * DIM);
  cvt_f32_f16<<<1024, 256, 0, stream>>>(c_proj_w, cpw16,  DIM * DIM);

  logmap_kernel<<<TOK, 256, 0, stream>>>(x, xhyp, ref_out);

  // QKV projection: [2048x1024] x [3072x1024]^T ; blocks = (M/16)*(N/512)
  gemm_wmma_f16f32<<<(TOK / 16) * (E3 / 512), 256, 0, stream>>>(
      xhyp, qkvw16, qkvf, TOK, E3, DIM);

  qkv_post_kernel<<<TOK, 256, 0, stream>>>(qkvf, q16, k16, vT16);

  flash_attn_kernel<<<(NHEAD * (TOK / 16)) / 8, 256, 0, stream>>>(
      q16, k16, vT16, y16);

  gemm_wmma_f16f32<<<(TOK / 16) * (DIM / 512), 256, 0, stream>>>(
      y16, cpw16, projf, TOK, DIM, DIM);

  expmap_kernel<<<TOK, 256, 0, stream>>>(projf, ref_out, y_out);
}
